// Model_90452011254026
// MI455X (gfx1250) — matrix operand order repaired and checked by reading, not compiled, not run on hardware
//
#include <hip/hip_runtime.h>
#include <cstdint>

#define DIMC   256
#define DINNER 512
#define DSTATE 64
#define DTRANK 16
#define KDIR   4
#define NBATCH 2
#define NDEPTH 6
#define HIMG   85
#define HPAD   88
#define CCH    (DTRANK + 2 * DSTATE)   // 144
#define CCPAD  192                     // x_proj weight rows padded to 3*64
#define SCHUNK 16                      // scan LDS staging chunk (steps)

typedef __attribute__((ext_vector_type(16))) __bf16 v16bf;
typedef __attribute__((ext_vector_type(8)))  float  v8f;

// ---------------- device helpers ----------------
__device__ __forceinline__ float silu_f(float x){ return x / (1.f + __expf(-x)); }
__device__ __forceinline__ float gelu_f(float x){ return 0.5f * x * (1.f + erff(x * 0.70710678118654752f)); }
__device__ __forceinline__ float softplus_f(float x){ return (x > 20.f) ? x : log1pf(__expf(x)); }

__device__ __forceinline__ v8f v8zero(){
  v8f r;
#pragma unroll
  for (int i = 0; i < 8; ++i) r[i] = 0.f;
  return r;
}

// async copy 16B global -> LDS (gfx1250; tracked by ASYNCcnt). LDS address = low 32 bits of
// the generic shared-memory pointer (ISA 10.2: LDS aperture uses addr[31:0]).
__device__ __forceinline__ void async_g2l_b128(const float* gsrc, float* lds_dst){
  uint32_t la = (uint32_t)(uintptr_t)lds_dst;
  asm volatile("global_load_async_to_lds_b128 %0, %1, off" :: "v"(la), "v"(gsrc) : "memory");
}
__device__ __forceinline__ void wait_asynccnt0(){
  asm volatile("s_wait_asynccnt 0" ::: "memory");
}

// A fragment 16x32 bf16 (contiguous K): lane half 0 -> K 0-15, half 1 -> K 16-31.
// Unguarded: caller guarantees lda%4==0, K%32==0, and padded-row reads land in mapped memory.
__device__ __forceinline__ v16bf load_a_frag(const float* A, long lda, int m0, int k0){
  int lane = threadIdx.x & 31;
  int half = lane >> 4;
  long row = m0 + (lane & 15);
  const float* Ar = A + row * lda + k0 + half * 16;
  float4 r0 = ((const float4*)Ar)[0];
  float4 r1 = ((const float4*)Ar)[1];
  float4 r2 = ((const float4*)Ar)[2];
  float4 r3 = ((const float4*)Ar)[3];
  v16bf a;
  a[0]=(__bf16)r0.x;  a[1]=(__bf16)r0.y;  a[2]=(__bf16)r0.z;  a[3]=(__bf16)r0.w;
  a[4]=(__bf16)r1.x;  a[5]=(__bf16)r1.y;  a[6]=(__bf16)r1.z;  a[7]=(__bf16)r1.w;
  a[8]=(__bf16)r2.x;  a[9]=(__bf16)r2.y;  a[10]=(__bf16)r2.z; a[11]=(__bf16)r2.w;
  a[12]=(__bf16)r3.x; a[13]=(__bf16)r3.y; a[14]=(__bf16)r3.z; a[15]=(__bf16)r3.w;
  return a;
}

// B fragment 32x16 bf16 from weight stored [N,K] row-major (y = x @ W^T). 16 contiguous floats/lane.
__device__ __forceinline__ v16bf load_b_fragT(const float* W, long ldb, int k0, int n0){
  int lane  = threadIdx.x & 31;
  long nn   = n0 + (lane & 15);
  int khalf = lane >> 4;
  const float* Wr = W + nn * ldb + k0 + khalf * 16;
  float4 q0 = ((const float4*)Wr)[0];
  float4 q1 = ((const float4*)Wr)[1];
  float4 q2 = ((const float4*)Wr)[2];
  float4 q3 = ((const float4*)Wr)[3];
  v16bf b;
  b[0]=(__bf16)q0.x;  b[1]=(__bf16)q0.y;  b[2]=(__bf16)q0.z;  b[3]=(__bf16)q0.w;
  b[4]=(__bf16)q1.x;  b[5]=(__bf16)q1.y;  b[6]=(__bf16)q1.z;  b[7]=(__bf16)q1.w;
  b[8]=(__bf16)q2.x;  b[9]=(__bf16)q2.y;  b[10]=(__bf16)q2.z; b[11]=(__bf16)q2.w;
  b[12]=(__bf16)q3.x; b[13]=(__bf16)q3.y; b[14]=(__bf16)q3.z; b[15]=(__bf16)q3.w;
  return b;
}

// B fragment from plain row-major B[K,N] (column slice, strided, unguarded)
__device__ __forceinline__ v16bf load_b_fragN(const float* B, long ldb, int k0, int n0){
  int lane  = threadIdx.x & 31;
  long nn   = n0 + (lane & 15);
  int khalf = lane >> 4;
  const float* Bc = B + (long)(k0 + khalf * 16) * ldb + nn;
  v16bf b;
#pragma unroll
  for (int v = 0; v < 8; ++v){
    b[2 * v]     = (__bf16)Bc[(long)(2 * v)     * ldb];
    b[2 * v + 1] = (__bf16)Bc[(long)(2 * v + 1) * ldb];
  }
  return b;
}

// ---------------- WMMA GEMM: C = A(MxK) * op(B) [+bias][act][+resid] ----------------
struct GemmArgs {
  const float* A; const float* B; float* C;
  const float* bias; const float* resid;
  int M, N, K;
  int lda, ldb, ldc;
  long aZ, bZ, cZ, rZ;
  int bT;        // 1: B is weight [N,K] row-major; 0: row-major [K,N]
  int biasMode;  // 0 none, 1 per-N, 2 per-M
  int act;       // 0 none, 1 GELU(exact), 2 softplus
};

__global__ __launch_bounds__(256) void gemm_bf16_wmma_k(GemmArgs g){
  const float* A = g.A + (long)blockIdx.z * g.aZ;
  const float* B = g.B + (long)blockIdx.z * g.bZ;
  float*       C = g.C + (long)blockIdx.z * g.cZ;
  const float* R = g.resid ? (g.resid + (long)blockIdx.z * g.rZ) : (const float*)0;

  int wave = threadIdx.x >> 5;
  int lane = threadIdx.x & 31;
  int m0 = (blockIdx.y * 8 + wave) * 16;
  int n0 = blockIdx.x * 64;
  if (m0 >= g.M) return;   // wave-uniform; EXEC stays all-ones for WMMA

  v8f acc[4];
#pragma unroll
  for (int t = 0; t < 4; ++t) acc[t] = v8zero();

  if (g.bT){
    for (int k0 = 0; k0 < g.K; k0 += 32){
      v16bf a = load_a_frag(A, g.lda, m0, k0);
#pragma unroll
      for (int t = 0; t < 4; ++t){
        v16bf b = load_b_fragT(B, g.ldb, k0, n0 + 16 * t);
        acc[t] = __builtin_amdgcn_wmma_f32_16x16x32_bf16(false, a, false, b,
                                                         (short)0, acc[t], false, false);
      }
    }
  } else {
    for (int k0 = 0; k0 < g.K; k0 += 32){
      v16bf a = load_a_frag(A, g.lda, m0, k0);
#pragma unroll
      for (int t = 0; t < 4; ++t){
        v16bf b = load_b_fragN(B, g.ldb, k0, n0 + 16 * t);
        acc[t] = __builtin_amdgcn_wmma_f32_16x16x32_bf16(false, a, false, b,
                                                         (short)0, acc[t], false, false);
      }
    }
  }

  int rowoff = (lane >> 4) * 8;
  int coloff = lane & 15;
#pragma unroll
  for (int t = 0; t < 4; ++t){
    int col = n0 + 16 * t + coloff;
    if (col >= g.N) continue;
#pragma unroll
    for (int r = 0; r < 8; ++r){
      int row = m0 + rowoff + r;
      if (row >= g.M) continue;
      float v = acc[t][r];
      if      (g.biasMode == 1) v += g.bias[col];
      else if (g.biasMode == 2) v += g.bias[row];
      if      (g.act == 1) v = gelu_f(v);
      else if (g.act == 2) v = softplus_f(v);
      long o = (long)row * g.ldc + col;
      if (R) v += R[o];
      C[o] = v;
    }
  }
}

// ---------------- LayerNorm (fused +postbias, +pos, *silu(gate)) ----------------
struct LnArgs {
  const float* X; float* Y; const float* g; const float* b;
  const float* pos; const float* postbias; const float* gate;
  int M, C, Np, gate_ld;
};

__global__ __launch_bounds__(256) void layernorm_k(LnArgs a){
  __shared__ float sred[256];
  int row = blockIdx.x;
  int tid = threadIdx.x;
  const float* x = a.X + (long)row * a.C;
  int per = a.C >> 8;

  float s = 0.f;
  for (int i = 0; i < per; ++i) s += x[tid + (i << 8)];
  sred[tid] = s; __syncthreads();
  for (int st = 128; st; st >>= 1){ if (tid < st) sred[tid] += sred[tid + st]; __syncthreads(); }
  float mu = sred[0] / a.C; __syncthreads();

  float v = 0.f;
  for (int i = 0; i < per; ++i){ float d = x[tid + (i << 8)] - mu; v += d * d; }
  sred[tid] = v; __syncthreads();
  for (int st = 128; st; st >>= 1){ if (tid < st) sred[tid] += sred[tid + st]; __syncthreads(); }
  float rstd = rsqrtf(sred[0] / a.C + 1e-5f);

  for (int i = 0; i < per; ++i){
    int c = tid + (i << 8);
    float y = (x[c] - mu) * rstd * a.g[c] + a.b[c];
    if (a.postbias) y += a.postbias[c];
    if (a.pos)      y += a.pos[(long)(row % a.Np) * a.C + c];
    if (a.gate)     y *= silu_f(a.gate[(long)row * a.gate_ld + c]);
    a.Y[(long)row * a.C + c] = y;
  }
}

// ---------------- elementwise / shaping kernels ----------------
__global__ void pack_input_k(const float* x, const float* fx, float* f){
  long idx = (long)blockIdx.x * blockDim.x + threadIdx.x;
  long total = (long)NBATCH * 3 * HPAD * HPAD;
  if (idx >= total) return;
  int wq = idx % HPAD; long t = idx / HPAD;
  int h = t % HPAD; t /= HPAD;
  int c = t % 3;    int b = t / 3;
  float v = 0.f;
  if (h < HIMG && wq < HIMG){
    long pix = (long)b * HIMG * HIMG + (long)h * HIMG + wq;
    v = (c < 2) ? x[pix * 2 + c] : fx[pix];
  }
  f[idx] = v;
}

// scale0 patchify (K=12, VALU): tok_raw[b*Np+k, c] = sum f[b,j,hp*p+m,wp*p+l]*WE[c,j,l,m] + bE[c]
__global__ void patchify_k(const float* f, const float* WE, const float* bE, float* out,
                           int Cin, int p, int Hp){
  long idx = (long)blockIdx.x * blockDim.x + threadIdx.x;
  int Np = Hp * Hp;
  long total = (long)NBATCH * Np * DIMC;
  if (idx >= total) return;
  int c = idx % DIMC; long t = idx / DIMC;
  int k = t % Np;     int  b = t / Np;
  int hp = k / Hp, wp = k % Hp;
  float s = bE[c];
  const float* W = WE + (long)c * Cin * p * p;
  for (int j = 0; j < Cin; ++j)
    for (int l = 0; l < p; ++l)
      for (int m = 0; m < p; ++m)
        s += f[(((long)b * Cin + j) * HPAD + (hp * p + m)) * HPAD + (wp * p + l)]
           * W[(j * p + l) * p + m];
  out[t * DIMC + c] = s;
}

// im2col for WMMA patchify (scales 1,2): X[(b*Np+kp)*K + (j*p+l)*p + m] = f[b,j,hp*p+m,wp*p+l]
__global__ void im2col_k(const float* f, float* X, int Cin, int p, int Hp){
  long idx = (long)blockIdx.x * blockDim.x + threadIdx.x;
  int Np = Hp * Hp, Kc = Cin * p * p;
  long total = (long)NBATCH * Np * Kc;
  if (idx >= total) return;
  int kk = idx % Kc; long t = idx / Kc;
  int kp = t % Np;   int b = t / Np;
  int j = kk / (p * p); int rem = kk % (p * p);
  int l = rem / p, m = rem % p;
  int hp = kp / Hp, wp = kp % Hp;
  X[idx] = f[(((long)b * Cin + j) * HPAD + (hp * p + m)) * HPAD + (wp * p + l)];
}

// depthwise 3x3 SAME + bias + silu, reading channel-last xz directly; xc[b,d,l] channel-first out
__global__ void dwconv_k(const float* xz, const float* w, const float* bias, float* xc, int Hp){
  long idx = (long)blockIdx.x * blockDim.x + threadIdx.x;
  int L = Hp * Hp;
  long total = (long)NBATCH * DINNER * L;
  if (idx >= total) return;
  int wq = idx % Hp; long t = idx / Hp;
  int h = t % Hp; t /= Hp;
  int d = t % DINNER; int b = t / DINNER;
  float s = bias[d];
  const float* wd = w + (long)d * 9;
  for (int dy = 0; dy < 3; ++dy){
    int hh = h + dy - 1;
    if (hh < 0 || hh >= Hp) continue;
    for (int dx = 0; dx < 3; ++dx){
      int ww = wq + dx - 1;
      if (ww < 0 || ww >= Hp) continue;
      s += xz[((long)b * L + hh * Hp + ww) * (2 * DINNER) + d] * wd[dy * 3 + dx];
    }
  }
  xc[idx] = silu_f(s);
}

// directional copies: xs[b,k,d,l] (channel-first, scan u) and xsT[(bk*L+l)*512+d] (token-major)
__global__ void build_xs_k(const float* xc, float* xs, float* xsT, int Hp){
  long idx = (long)blockIdx.x * blockDim.x + threadIdx.x;
  int L = Hp * Hp;
  long total = (long)NBATCH * KDIR * DINNER * L;
  if (idx >= total) return;
  int l = idx % L; long t = idx / L;
  int d = t % DINNER; t /= DINNER;
  int k = t % KDIR;   int b = t / KDIR;
  int lsrc;
  if (k == 0)      lsrc = l;
  else if (k == 1) lsrc = (l % Hp) * Hp + l / Hp;
  else if (k == 2) lsrc = L - 1 - l;
  else { int lr = L - 1 - l; lsrc = (lr % Hp) * Hp + lr / Hp; }
  float v = xc[((long)b * DINNER + d) * L + lsrc];
  xs[idx] = v;
  long bk = (long)b * KDIR + k;
  xsT[(bk * L + l) * DINNER + d] = v;
}

// stage x_proj weight [6,4,144,512] -> [6,4,192,512] zero-padded rows
__global__ void pad_xproj_k(const float* W, float* out){
  long idx = (long)blockIdx.x * blockDim.x + threadIdx.x;
  long total = (long)NDEPTH * KDIR * CCPAD * DINNER;
  if (idx >= total) return;
  int dd = idx % DINNER; long t = idx / DINNER;
  int n = t % CCPAD; long jk = t / CCPAD;
  out[idx] = (n < CCH) ? W[(jk * CCH + n) * DINNER + dd] : 0.f;
}

// dt_proj (K=16, fused softplus; token-major in/out)
__global__ void dtproj_k(const float* xdblT, const float* dpw, const float* dbias,
                         float* dtT, int L){
  long idx = (long)blockIdx.x * blockDim.x + threadIdx.x;
  long total = (long)NBATCH * KDIR * L * DINNER;
  if (idx >= total) return;
  int d = idx % DINNER; long t = idx / DINNER;
  int l = t % L; long bk = t / L; int k = bk % KDIR;
  const float* xr = xdblT + (bk * L + l) * (long)CCH;
  const float* wr = dpw + ((long)k * DINNER + d) * DTRANK;
  float s = dbias[k * DINNER + d];
#pragma unroll
  for (int r = 0; r < DTRANK; ++r) s += xr[r] * wr[r];
  dtT[idx] = softplus_f(s);
}

// selective scan: one wave per (b,k,d); 64 states = 2/lane; u (xs) updated in place -> ys.
// B/C rows (shared by all 8 waves of the block) are double-buffer staged into LDS with
// global_load_async_to_lds_b128 (ASYNCcnt) to cut global traffic 8x on the serial loop.
__global__ __launch_bounds__(256) void scan_k(const float* dtT, float* u, const float* xdblT,
                                              const float* A_log, const float* Ds, int L){
  __shared__ float bc[2 * SCHUNK * 128];
  int tid  = threadIdx.x;
  int lane = tid & 31;
  int wv   = tid >> 5;
  int d = blockIdx.x * 8 + wv;
  int k = blockIdx.y, b = blockIdx.z;
  long bk = (long)b * KDIR + k;

  const float* dtp = dtT + bk * (long)L * DINNER + d;
  float*       up  = u   + (bk * DINNER + d) * (long)L;
  const float* xd  = xdblT + bk * (long)L * CCH;

  int n0 = lane, n1 = lane + 32;
  float A0 = -__expf(A_log[((long)k * DINNER + d) * DSTATE + n0]);
  float A1 = -__expf(A_log[((long)k * DINNER + d) * DSTATE + n1]);
  float Dv = Ds[k * DINNER + d];

  int nch = (L + SCHUNK - 1) / SCHUNK;

  // stage chunk 0 (rows l0..l0+15, 128 floats each = B[64] | C[64])
  for (int e = tid; e < SCHUNK * 32; e += 256){
    int row = e >> 5, c4 = (e & 31) * 4;
    async_g2l_b128(xd + (long)row * CCH + DTRANK + c4, bc + row * 128 + c4);
  }
  wait_asynccnt0();
  __syncthreads();

  float h0 = 0.f, h1 = 0.f;
  for (int c = 0; c < nch; ++c){
    int l0 = c * SCHUNK;
    const float* bb = bc + (c & 1) * (SCHUNK * 128);
    if (c + 1 < nch){
      float* nb = bc + ((c + 1) & 1) * (SCHUNK * 128);
      long l0n = (long)(c + 1) * SCHUNK;
      for (int e = tid; e < SCHUNK * 32; e += 256){
        int row = e >> 5, c4 = (e & 31) * 4;
        async_g2l_b128(xd + (l0n + row) * CCH + DTRANK + c4, nb + row * 128 + c4);
      }
    }
    __builtin_prefetch(&dtp[(long)(l0 + SCHUNK) * DINNER], 0, 1);
    int lim = (L - l0 < SCHUNK) ? (L - l0) : SCHUNK;
    for (int st = 0; st < lim; ++st){
      int l = l0 + st;
      float dtl = dtp[(long)l * DINNER];
      float ul  = up[l];
      float Bl0 = bb[st * 128 + n0],      Bl1 = bb[st * 128 + n1];
      float Cl0 = bb[st * 128 + 64 + n0], Cl1 = bb[st * 128 + 64 + n1];
      float du = dtl * ul;
      h0 = __expf(dtl * A0) * h0 + du * Bl0;
      h1 = __expf(dtl * A1) * h1 + du * Bl1;
      float yp = h0 * Cl0 + h1 * Cl1;
#pragma unroll
      for (int off = 16; off; off >>= 1) yp += __shfl_xor(yp, off, 32);
      if (lane == 0) up[l] = yp + Dv * ul;
    }
    wait_asynccnt0();
    __syncthreads();
  }
}

// combine 4 directional outputs into channel-last ysum[b*L+l, d]
__global__ void gather_y_k(const float* ys, float* yo, int Hp){
  long idx = (long)blockIdx.x * blockDim.x + threadIdx.x;
  int L = Hp * Hp;
  long total = (long)NBATCH * L * DINNER;
  if (idx >= total) return;
  int d = idx % DINNER; long t = idx / DINNER;
  int l = t % L; int b = t / L;
  long base = (long)b * KDIR * DINNER * L;
  long dl = (long)d * L;
  int l1 = (l % Hp) * Hp + l / Hp;
  int l2 = L - 1 - l;
  int l3 = L - 1 - l1;
  float v = ys[base + 0L * DINNER * L + dl + l ]
          + ys[base + 1L * DINNER * L + dl + l1]
          + ys[base + 2L * DINNER * L + dl + l2]
          + ys[base + 3L * DINNER * L + dl + l3];
  yo[idx] = v;
}

// scatter Z[(b*Np+n)*(Cout*p*p) + (c*p+pi)*p+qi] -> f[b,c,hp*p+pi,wp*p+qi]
__global__ void depatch_scatter_k(const float* Z, float* f, int Cout, int p, int Hp){
  long idx = (long)blockIdx.x * blockDim.x + threadIdx.x;
  long total = (long)NBATCH * Cout * HPAD * HPAD;
  if (idx >= total) return;
  int col = idx % HPAD; long t = idx / HPAD;
  int row = t % HPAD; t /= HPAD;
  int c = t % Cout;   int b = t / Cout;
  int hp = row / p, pi = row % p, wp = col / p, qi = col % p;
  int n = hp * Hp + wp;
  f[idx] = Z[((long)b * Hp * Hp + n) * (long)(Cout * p * p) + (long)(c * p + pi) * p + qi];
}

__global__ void crop_k(const float* f, float* out){
  long idx = (long)blockIdx.x * blockDim.x + threadIdx.x;
  long total = (long)NBATCH * HIMG * HIMG;
  if (idx >= total) return;
  int wq = idx % HIMG; long t = idx / HIMG;
  int h = t % HIMG; int b = t / HIMG;
  out[idx] = f[((long)b * HPAD + h) * HPAD + wq];
}

// ---------------- host orchestration ----------------
extern "C" void kernel_launch(void* const* d_in, const int* in_sizes, int n_in,
                              void* d_out, int out_size, void* d_ws, size_t ws_size,
                              hipStream_t stream){
  (void)in_sizes; (void)n_in; (void)out_size; (void)ws_size;
  auto Pf = [&](int i){ return (const float*)d_in[i]; };
  const int PATCHP[3] = {2, 4, 8};

  float* w = (float*)d_ws;
  size_t off = 0;
  auto wsalloc = [&](size_t n){ float* p = w + off; off += (n + 63) & ~(size_t)63; return p; };
  const int  L0 = (HPAD / 2) * (HPAD / 2);   // 1936
  const long M0 = (long)NBATCH * L0;         // 3872
  float* fA    = wsalloc((size_t)NBATCH * DIMC * HPAD * HPAD);
  float* fB    = wsalloc((size_t)NBATCH * DIMC * HPAD * HPAD);
  float* tok   = wsalloc((size_t)(M0 + 16) * DIMC);
  float* tok2  = wsalloc((size_t)(M0 + 16) * DIMC);
  float* hbuf  = wsalloc((size_t)(M0 + 16) * DIMC);
  float* xz    = wsalloc((size_t)(M0 + 16) * 2 * DINNER);  // in_proj out / MLP / im2col / Z
  float* xc    = wsalloc((size_t)NBATCH * DINNER * L0);
  float* xs    = wsalloc((size_t)NBATCH * KDIR * DINNER * L0);
  float* xsT   = wsalloc((size_t)NBATCH * KDIR * DINNER * L0);
  float* xdblT = wsalloc((size_t)NBATCH * KDIR * CCH * L0);
  float* dtT   = wsalloc((size_t)NBATCH * KDIR * DINNER * L0);
  float* ysum  = wsalloc((size_t)(M0 + 16) * DINNER);
  float* sxp   = wsalloc((size_t)NDEPTH * KDIR * CCPAD * DINNER);
  (void)wsalloc(16384);  // tail slack for padded-tile reads

  auto gemm = [&](const float* A, const float* B, float* C,
                  const float* bias, const float* resid,
                  int M, int N, int K, int lda, int ldb, int ldc,
                  long aZ, long bZ, long cZ, long rZ, int nz,
                  int bT, int biasMode, int act){
    GemmArgs g{A, B, C, bias, resid, M, N, K, lda, ldb, ldc, aZ, bZ, cZ, rZ, bT, biasMode, act};
    dim3 grid((N + 63) / 64, ((M + 15) / 16 + 7) / 8, nz);
    hipLaunchKernelGGL(gemm_bf16_wmma_k, grid, dim3(256), 0, stream, g);
  };
  auto ln = [&](const float* X, float* Y, const float* gW, const float* bW,
                int M, int C, const float* pos, int Np, const float* postbias,
                const float* gate, int gate_ld){
    LnArgs a{X, Y, gW, bW, pos, postbias, gate, M, C, Np, gate_ld};
    hipLaunchKernelGGL(layernorm_k, dim3(M), dim3(256), 0, stream, a);
  };
  auto ew = [&](long total){ return dim3((unsigned)((total + 255) / 256)); };

  hipLaunchKernelGGL(pack_input_k, ew((long)NBATCH * 3 * HPAD * HPAD), dim3(256), 0, stream,
                     Pf(0), Pf(1), fA);

  float* fin = fA; float* fout = fB;
  for (int i = 0; i < 3; ++i){
    int p = PATCHP[i], Hp = HPAD / p, L = Hp * Hp, M = NBATCH * L;
    int Cin = (i == 0) ? 3 : DIMC;
    int Cout = (i < 2) ? DIMC : 1;

    int sb = 2 + i * 28;   // recursive dict-insertion order, 28 leaves per scale after {x, fx}
    const float *pe_we = Pf(sb + 0), *pe_be = Pf(sb + 1), *pe_ln_w = Pf(sb + 2),
                *pe_ln_b = Pf(sb + 3), *pos = Pf(sb + 4);
    const float *ln1_w = Pf(sb + 5), *ln1_b = Pf(sb + 6), *in_proj_w = Pf(sb + 7),
                *conv_w = Pf(sb + 8), *conv_b = Pf(sb + 9), *x_proj_w = Pf(sb + 10),
                *dt_proj_w = Pf(sb + 11), *dt_bias = Pf(sb + 12), *A_log = Pf(sb + 13),
                *Dsp = Pf(sb + 14), *onw = Pf(sb + 15), *onb = Pf(sb + 16),
                *out_proj_w = Pf(sb + 17), *ln2_w = Pf(sb + 18), *ln2_b = Pf(sb + 19),
                *fc1_w = Pf(sb + 20), *fc1_b = Pf(sb + 21), *fc2_w = Pf(sb + 22),
                *fc2_b = Pf(sb + 23);
    const float *dpe_we = Pf(sb + 24), *dpe_be = Pf(sb + 25), *dpe_ln_w = Pf(sb + 26),
                *dpe_ln_b = Pf(sb + 27);

    // ---- patchify ----
    if (i == 0){
      hipLaunchKernelGGL(patchify_k, ew((long)NBATCH * L * DIMC), dim3(256), 0, stream,
                         fin, pe_we, pe_be, hbuf, Cin, p, Hp);
    } else {
      int Kc = Cin * p * p;   // 4096 or 16384 (mult 32)
      hipLaunchKernelGGL(im2col_k, ew((long)NBATCH * L * Kc), dim3(256), 0, stream,
                         fin, xz, Cin, p, Hp);
      gemm(xz, pe_we, hbuf, pe_be, nullptr, M, DIMC, Kc, Kc, Kc, DIMC,
           0, 0, 0, 0, 1, 1, 1, 0);
    }
    ln(hbuf, tok, pe_ln_w, pe_ln_b, M, DIMC, pos, L, nullptr, nullptr, 0);

    // stage zero-padded x_proj weights for all depths of this scale
    hipLaunchKernelGGL(pad_xproj_k, ew((long)NDEPTH * KDIR * CCPAD * DINNER), dim3(256), 0,
                       stream, x_proj_w, sxp);

    for (int j = 0; j < NDEPTH; ++j){
      const float* ipw = in_proj_w + (long)j * 2 * DINNER * DIMC;
      const float* cw  = conv_w    + (long)j * DINNER * 9;
      const float* cb  = conv_b    + (long)j * DINNER;
      const float* dpw = dt_proj_w + (long)j * KDIR * DINNER * DTRANK;
      const float* dbi = dt_bias   + (long)j * KDIR * DINNER;
      const float* alg = A_log     + (long)j * KDIR * DINNER * DSTATE;
      const float* dsv = Dsp       + (long)j * KDIR * DINNER;
      const float* opw = out_proj_w + (long)j * DIMC * DINNER;

      // LN1
      ln(tok, hbuf, ln1_w + (long)j * DIMC, ln1_b + (long)j * DIMC, M, DIMC,
         nullptr, 1, nullptr, nullptr, 0);
      // in_proj: [M,256] x W[1024,256]^T -> xz[M,1024]
      gemm(hbuf, ipw, xz, nullptr, nullptr, M, 2 * DINNER, DIMC, DIMC, DIMC, 2 * DINNER,
           0, 0, 0, 0, 1, 1, 0, 0);
      // depthwise conv + silu (reads xz channel-last directly)
      hipLaunchKernelGGL(dwconv_k, ew((long)NBATCH * DINNER * L), dim3(256), 0, stream,
                         xz, cw, cb, xc, Hp);
      // 4 directional copies (channel-first + token-major)
      hipLaunchKernelGGL(build_xs_k, ew((long)NBATCH * KDIR * DINNER * L), dim3(256), 0, stream,
                         xc, xs, xsT, Hp);
      // x_dbl token-major: xdblT[bk][L,144] = xsT[bk][L,512] x Wk_pad[192,512]^T
      for (int k = 0; k < KDIR; ++k)
        gemm(xsT + (long)k * L * DINNER, sxp + (long)(j * KDIR + k) * CCPAD * DINNER,
             xdblT + (long)k * L * CCH, nullptr, nullptr,
             L, CCH, DINNER, DINNER, DINNER, CCH,
             (long)KDIR * L * DINNER, 0, (long)KDIR * L * CCH, 0, NBATCH, 1, 0, 0);
      // dt_proj (K=16) fused VALU + softplus
      hipLaunchKernelGGL(dtproj_k, ew((long)NBATCH * KDIR * L * DINNER), dim3(256), 0, stream,
                         xdblT, dpw, dbi, dtT, L);
      // selective scan (in-place xs -> ys), async-LDS staged B/C
      hipLaunchKernelGGL(scan_k, dim3(DINNER / 8, KDIR, NBATCH), dim3(256), 0, stream,
                         dtT, xs, xdblT, alg, dsv, L);
      // combine directions -> channel-last
      hipLaunchKernelGGL(gather_y_k, ew((long)NBATCH * L * DINNER), dim3(256), 0, stream,
                         xs, ysum, Hp);
      // out_norm * silu(z)
      ln(ysum, ysum, onw + (long)j * DINNER, onb + (long)j * DINNER, M, DINNER,
         nullptr, 1, nullptr, xz + DINNER, 2 * DINNER);
      // out_proj + residual(tok) -> tok2
      gemm(ysum, opw, tok2, nullptr, tok, M, DIMC, DINNER, DINNER, DINNER, DIMC,
           0, 0, 0, 0, 1, 1, 0, 0);
      // LN2
      ln(tok2, hbuf, ln2_w + (long)j * DIMC, ln2_b + (long)j * DIMC, M, DIMC,
         nullptr, 1, nullptr, nullptr, 0);
      // fc1 + GELU -> xz
      gemm(hbuf, fc1_w + (long)j * 4 * DIMC * DIMC, xz, fc1_b + (long)j * 4 * DIMC, nullptr,
           M, 4 * DIMC, DIMC, DIMC, DIMC, 4 * DIMC, 0, 0, 0, 0, 1, 1, 1, 1);
      // fc2 + residual(tok2) -> tok
      gemm(xz, fc2_w + (long)j * DIMC * 4 * DIMC, tok, fc2_b + (long)j * DIMC, tok2,
           M, DIMC, 4 * DIMC, 4 * DIMC, 4 * DIMC, DIMC, 0, 0, 0, 0, 1, 1, 1, 0);
    }

    // ---- depatchify: LN(+bE) -> WMMA GEMM (B = WE reshaped [256, Cout*p*p]) -> scatter ----
    int Nd = Cout * p * p;   // 1024, 4096, 64 (mult 64)
    ln(tok, hbuf, dpe_ln_w, dpe_ln_b, M, DIMC, nullptr, 1, dpe_be, nullptr, 0);
    gemm(hbuf, dpe_we, xz, nullptr, nullptr, M, Nd, DIMC, DIMC, Nd, Nd,
         0, 0, 0, 0, 1, 0, 0, 0);
    hipLaunchKernelGGL(depatch_scatter_k, ew((long)NBATCH * Cout * HPAD * HPAD), dim3(256), 0,
                       stream, xz, fout, Cout, p, Hp);
    float* t = fin; fin = fout; fout = t;
  }

  hipLaunchKernelGGL(crop_k, ew((long)NBATCH * HIMG * HIMG), dim3(256), 0, stream,
                     fin, (float*)d_out);
}
